// GPT2Model_23536420782247
// MI455X (gfx1250) — matrix operand order repaired and checked by reading, not compiled, not run on hardware
//
#include <hip/hip_runtime.h>
#include <hip/hip_bf16.h>

typedef __bf16 bf16_t;
typedef __attribute__((ext_vector_type(16))) __bf16 v16bf;
typedef __attribute__((ext_vector_type(8)))  float  v8f;

#define Dm   768
#define Hh   12
#define DHd  64
#define Ll   8
#define Bb   4
#define Ss   1024
#define FFf  3072
#define Vv   50257
#define Mm   (Bb*Ss)   /* 4096 rows of activations */

enum { F_OUT_BF16 = 1, F_TRANS = 2, F_RELU = 4, F_CAUSAL = 8, F_KCAUSAL = 16, F_NARROW = 32 };

union FragU { v16bf v; uint4 q[2]; };

// A-matrix 16x32 bf16 fragment (ISA 7.12.2): lanes 0-15 row M=l, K=0..7 & 16..23;
// lanes 16-31 row M=l-16, K=8..15 & 24..31.  Two 16B loads per lane.
__device__ __forceinline__ v16bf load_a_frag(const bf16_t* row, int k0, int hi) {
    const uint4* p = reinterpret_cast<const uint4*>(row + k0 + (hi << 3));
    FragU f; f.q[0] = p[0]; f.q[1] = p[2]; return f.v;
}
// B-matrix 32x16 bf16 fragment: lanes 0-15 col N=l, K=0..7 & 16..23; lanes 16-31 K=8..15 & 24..31.
// Two 16B loads per lane (B supplied as Bt[N,K] row-major).
__device__ __forceinline__ v16bf load_b_frag(const bf16_t* row, int k0, int hi) {
    const uint4* p = reinterpret_cast<const uint4*>(row + k0 + (hi << 3));
    FragU f; f.q[0] = p[0]; f.q[1] = p[2]; return f.v;
}

struct Tile { v16bf a0, a1, b[4]; };

__device__ __forceinline__ void load_tile(Tile& t, const bf16_t* aRow0, const bf16_t* aRow1,
                                          const bf16_t* const* bRow, int k0, int hi) {
    t.a0 = load_a_frag(aRow0, k0, hi);
    t.a1 = load_a_frag(aRow1, k0, hi);
#pragma unroll
    for (int i = 0; i < 4; ++i) t.b[i] = load_b_frag(bRow[i], k0, hi);
}

__device__ __forceinline__ void mma_tile(const Tile& t, v8f acc[2][4]) {
#pragma unroll
    for (int i = 0; i < 4; ++i) {
        acc[0][i] = __builtin_amdgcn_wmma_f32_16x16x32_bf16(
            false, t.a0, false, t.b[i], (short)0, acc[0][i], false, false);
        acc[1][i] = __builtin_amdgcn_wmma_f32_16x16x32_bf16(
            false, t.a1, false, t.b[i], (short)0, acc[1][i], false, false);
    }
}

// ---------------- generic batched bf16 WMMA GEMM --------------------------
// C[z][M,N] = A[z][M,K] * Bt[z][N,K]^T  (+bias)(+residual)(relu)(scale)
// Per-wave tile 32x64 (8 WMMAs / 32-K slab); block tile 128x128 (8 waves)
// or 256x64 in F_NARROW mode (for N=64 GEMMs).  Register ping-pong double
// buffering so VMEM overlaps the matrix pipe.  F_KCAUSAL clips the K loop to
// mTile+32 (causal attention P*V: probabilities are zero for k > q).
// z offsets use two strides: off = (z/zInner)*Outer + (z%zInner)*Inner
__global__ __launch_bounds__(256)
void gemm_bf16_wmma(const bf16_t* __restrict__ A, long long aO, long long aI, int lda,
                    const bf16_t* __restrict__ Bt, long long bO, long long bI, int ldb,
                    void* Cv, long long cO, long long cI, int ldc,
                    int M, int N, int K, int zInner,
                    const float* __restrict__ bias, const float* residual,
                    float scale, int flags) {
    const int z = blockIdx.z;
    const long long za = (long long)(z / zInner) * aO + (long long)(z % zInner) * aI;
    const long long zb = (long long)(z / zInner) * bO + (long long)(z % zInner) * bI;
    const long long zc = (long long)(z / zInner) * cO + (long long)(z % zInner) * cI;

    const int lane = threadIdx.x & 31;
    const int wave = threadIdx.x >> 5;
    int mTile, nBase;
    if (flags & F_NARROW) {             // 8x1 wave layout, block tile 256x64
        mTile = blockIdx.y * 256 + wave * 32;
        nBase = blockIdx.x * 64;
    } else {                            // 4x2 wave layout, block tile 128x128
        mTile = blockIdx.y * 128 + (wave & 3) * 32;
        nBase = blockIdx.x * 128 + (wave >> 2) * 64;
    }
    if ((flags & F_CAUSAL) && (nBase > mTile + 31)) return;  // fully-masked causal tile

    const int r16 = lane & 15;
    const int hi  = lane >> 4;

    const bf16_t* aRow0 = A + za + (long long)(mTile + r16) * lda;
    const bf16_t* aRow1 = aRow0 + (long long)16 * lda;
    const bf16_t* bRow[4];
#pragma unroll
    for (int i = 0; i < 4; ++i) {
        int n  = nBase + i * 16 + r16;
        int nc = (n < N) ? n : 0;               // clamp for ragged N (lm_head)
        bRow[i] = Bt + zb + (long long)nc * ldb;
    }

    // causal P*V: rows [mTile, mTile+32) have zero probs for k >= mTile+32
    const int Keff = (flags & F_KCAUSAL) ? ((mTile + 32 < K) ? mTile + 32 : K) : K;

    v8f acc[2][4] = {};
    Tile t0, t1;
    load_tile(t0, aRow0, aRow1, bRow, 0, hi);
    for (int k0 = 0; k0 < Keff; k0 += 64) {      // Keff is a multiple of 32
        const bool has2 = (k0 + 32 < Keff);
        if (has2) load_tile(t1, aRow0, aRow1, bRow, k0 + 32, hi);
        if (k0 + 128 < Keff) {                   // global_prefetch_b8 two slabs ahead
            __builtin_prefetch(aRow0 + k0 + 128, 0, 1);
            __builtin_prefetch(bRow[0] + k0 + 128, 0, 1);
            __builtin_prefetch(bRow[2] + k0 + 128, 0, 1);
        }
        mma_tile(t0, acc);
        if (k0 + 64 < Keff) load_tile(t0, aRow0, aRow1, bRow, k0 + 64, hi);
        if (has2) mma_tile(t1, acc);
    }

    // C fragment: VGPR r, lanes0-15 -> M=r, lanes16-31 -> M=8+r; N = lane%16
#pragma unroll
    for (int ms = 0; ms < 2; ++ms) {
#pragma unroll
        for (int i = 0; i < 4; ++i) {
            int n = nBase + i * 16 + r16;
            if (n >= N) continue;
            int mBase = mTile + ms * 16 + hi * 8;
            if (flags & F_TRANS) {               // bf16 store transposed: out[n, mBase..+8)
                union { bf16_t h[8]; uint4 u; } pk;
#pragma unroll
                for (int r = 0; r < 8; ++r) pk.h[r] = (bf16_t)(acc[ms][i][r] * scale);
                *reinterpret_cast<uint4*>((bf16_t*)Cv + zc + (long long)n * ldc + mBase) = pk.u;
            } else {
#pragma unroll
                for (int r = 0; r < 8; ++r) {
                    int m = mBase + r;
                    float v = acc[ms][i][r] * scale;
                    if (bias)     v += bias[n];
                    if (residual) v += residual[(long long)m * ldc + n];
                    if (flags & F_RELU) v = fmaxf(v, 0.0f);
                    long long off = zc + (long long)m * ldc + n;
                    if (flags & F_OUT_BF16) ((bf16_t*)Cv)[off] = (bf16_t)v;
                    else                    ((float*)Cv)[off]  = v;
                }
            }
        }
    }
}

// ---------------- wave32 block reductions ---------------------------------
__device__ __forceinline__ float blockReduceSum(float v) {
    __shared__ float sh[8];
    int lane = threadIdx.x & 31, w = threadIdx.x >> 5;
#pragma unroll
    for (int o = 16; o; o >>= 1) v += __shfl_down(v, o, 32);
    if (lane == 0) sh[w] = v;
    __syncthreads();
    if (w == 0) {
        float t = (lane < 8) ? sh[lane] : 0.0f;
#pragma unroll
        for (int o = 4; o; o >>= 1) t += __shfl_down(t, o, 32);
        if (lane == 0) sh[0] = t;
    }
    __syncthreads();
    float r = sh[0];
    __syncthreads();
    return r;
}
__device__ __forceinline__ float blockReduceMax(float v) {
    __shared__ float sh[8];
    int lane = threadIdx.x & 31, w = threadIdx.x >> 5;
#pragma unroll
    for (int o = 16; o; o >>= 1) v = fmaxf(v, __shfl_down(v, o, 32));
    if (lane == 0) sh[w] = v;
    __syncthreads();
    if (w == 0) {
        float t = (lane < 8) ? sh[lane] : -1e30f;
#pragma unroll
        for (int o = 4; o; o >>= 1) t = fmaxf(t, __shfl_down(t, o, 32));
        if (lane == 0) sh[0] = t;
    }
    __syncthreads();
    float r = sh[0];
    __syncthreads();
    return r;
}

// ---------------- small kernels -------------------------------------------
__global__ __launch_bounds__(256)
void embed_kernel(const int* __restrict__ idx, const float* __restrict__ tok,
                  const float* __restrict__ pos, float* __restrict__ x, int total) {
    int i = blockIdx.x * 256 + threadIdx.x;
    if (i >= total) return;
    int row = i / Dm, c = i - row * Dm;
    x[i] = tok[(long long)idx[row] * Dm + c] + pos[(long long)(row % Ss) * Dm + c];
}

__global__ __launch_bounds__(256)
void convert_transpose(const float* __restrict__ in, bf16_t* __restrict__ out,
                       int K, int N) {
    long long i = (long long)blockIdx.x * 256 + threadIdx.x;
    if (i >= (long long)K * N) return;
    int k = (int)(i / N), n = (int)(i - (long long)k * N);
    out[(long long)n * K + k] = (bf16_t)in[i];
}

__global__ __launch_bounds__(256)
void layernorm_bf16(const float* __restrict__ x, const float* __restrict__ g,
                    const float* __restrict__ b, bf16_t* __restrict__ out) {
    long long row = blockIdx.x;
    const float* xr = x + row * Dm;
    float s = 0.f, s2 = 0.f;
    for (int c = threadIdx.x; c < Dm; c += 256) { float v = xr[c]; s += v; s2 += v * v; }
    s  = blockReduceSum(s);
    s2 = blockReduceSum(s2);
    float mean = s / Dm;
    float var  = s2 / Dm - mean * mean;
    float rstd = rsqrtf(var + 1e-5f);
    for (int c = threadIdx.x; c < Dm; c += 256)
        out[row * Dm + c] = (bf16_t)((xr[c] - mean) * rstd * g[c] + b[c]);
}

// causal softmax; writes only columns [0, (q&~31)+32) — exactly the region
// the K-clipped P*V GEMM reads (zero tail beyond the row's 32-aligned tile).
__global__ __launch_bounds__(256)
void softmax_causal(const float* __restrict__ scores, bf16_t* __restrict__ att) {
    long long row = blockIdx.x;            // over B*H*S
    int q = (int)(row % Ss);
    int valid = q + 1;
    int lim = (q & ~31) + 32;
    const float* sr = scores + row * Ss;
    bf16_t* ar = att + row * Ss;
    float mx = -1e30f;
    for (int c = threadIdx.x; c < valid; c += 256) mx = fmaxf(mx, sr[c]);
    mx = blockReduceMax(mx);
    float sum = 0.f;
    for (int c = threadIdx.x; c < valid; c += 256) sum += __expf(sr[c] - mx);
    sum = blockReduceSum(sum);
    float inv = 1.0f / sum;
    for (int c = threadIdx.x; c < lim; c += 256)
        ar[c] = (c < valid) ? (bf16_t)(__expf(sr[c] - mx) * inv) : (bf16_t)0.0f;
}

__global__ __launch_bounds__(256)
void loss_rows(const float* __restrict__ logits, const int* __restrict__ tgt,
               float* __restrict__ nll) {
    long long row = blockIdx.x;
    const float* lr = logits + row * (long long)Vv;
    float mx = -1e30f;
    for (int c = threadIdx.x; c < Vv; c += 256) mx = fmaxf(mx, lr[c]);
    mx = blockReduceMax(mx);
    float sum = 0.f;
    for (int c = threadIdx.x; c < Vv; c += 256) sum += __expf(lr[c] - mx);
    sum = blockReduceSum(sum);
    if (threadIdx.x == 0) nll[row] = mx + logf(sum) - lr[tgt[row]];
}

__global__ __launch_bounds__(256)
void loss_mean(const float* __restrict__ nll, float* __restrict__ out, int n) {
    float s = 0.f;
    for (int i = threadIdx.x; i < n; i += 256) s += nll[i];
    s = blockReduceSum(s);
    if (threadIdx.x == 0) out[0] = s / n;
}

// ---------------- host-side orchestration --------------------------------
static inline unsigned cdiv(long long a, long long b) { return (unsigned)((a + b - 1) / b); }

extern "C" void kernel_launch(void* const* d_in, const int* in_sizes, int n_in,
                              void* d_out, int out_size, void* d_ws, size_t ws_size,
                              hipStream_t stream) {
    (void)in_sizes; (void)n_in; (void)out_size; (void)ws_size;
    const float* tok  = (const float*)d_in[0];
    const float* pos  = (const float*)d_in[1];
    const float* Wq   = (const float*)d_in[2];
    const float* Wk   = (const float*)d_in[3];
    const float* Wv   = (const float*)d_in[4];
    const float* Wo   = (const float*)d_in[5];
    const float* ln1g = (const float*)d_in[6];
    const float* ln1b = (const float*)d_in[7];
    const float* ln2g = (const float*)d_in[8];
    const float* ln2b = (const float*)d_in[9];
    const float* W1   = (const float*)d_in[10];
    const float* b1   = (const float*)d_in[11];
    const float* W2   = (const float*)d_in[12];
    const float* b2   = (const float*)d_in[13];
    const float* lnfg = (const float*)d_in[14];
    const float* lnfb = (const float*)d_in[15];
    const float* lmh  = (const float*)d_in[16];
    const int*   idx  = (const int*)d_in[17];
    const int*   tgt  = (const int*)d_in[18];

    float* logits  = (float*)d_out;
    float* lossOut = logits + (size_t)Mm * Vv;

    char* base = (char*)d_ws;
    size_t off = 0;
    auto take = [&](size_t bytes) -> char* {
        char* p = base + off; off = (off + bytes + 255) & ~(size_t)255; return p;
    };
    bf16_t* WqT  = (bf16_t*)take((size_t)Ll * Dm * Dm * 2);
    bf16_t* WkT  = (bf16_t*)take((size_t)Ll * Dm * Dm * 2);
    bf16_t* WvT  = (bf16_t*)take((size_t)Ll * Dm * Dm * 2);
    bf16_t* WoT  = (bf16_t*)take((size_t)Ll * Dm * Dm * 2);
    bf16_t* W1T  = (bf16_t*)take((size_t)Ll * Dm * FFf * 2);
    bf16_t* W2T  = (bf16_t*)take((size_t)Ll * Dm * FFf * 2);
    bf16_t* lmhT = (bf16_t*)take((size_t)Dm * Vv * 2);
    float*  x    = (float*) take((size_t)Mm * Dm * 4);
    bf16_t* hbf  = (bf16_t*)take((size_t)Mm * Dm * 2);
    bf16_t* qbf  = (bf16_t*)take((size_t)Mm * Dm * 2);
    bf16_t* kbf  = (bf16_t*)take((size_t)Mm * Dm * 2);
    bf16_t* vtbf = (bf16_t*)take((size_t)Dm * Mm * 2);   // [768, 4096] = Vᵀ
    bf16_t* obf  = (bf16_t*)take((size_t)Mm * Dm * 2);
    bf16_t* ffbf = (bf16_t*)take((size_t)Mm * FFf * 2);
    float*  sc   = (float*) take((size_t)Bb * Hh * Ss * Ss * 4);
    bf16_t* att  = (bf16_t*)take((size_t)Bb * Hh * Ss * Ss * 2);
    float*  nll  = (float*) take((size_t)Mm * 4);

    dim3 blk(256);
    auto gemm = [&](const bf16_t* A, long long aO, long long aI, int lda,
                    const bf16_t* Bt, long long bO, long long bI, int ldb,
                    void* C, long long cO, long long cI, int ldc,
                    int M, int N, int K, int z, int zInner,
                    const float* bias, const float* res, float scale, int flags) {
        dim3 g((flags & F_NARROW) ? cdiv(N, 64)  : cdiv(N, 128),
               (flags & F_NARROW) ? (unsigned)(M / 256) : (unsigned)(M / 128),
               (unsigned)z);
        gemm_bf16_wmma<<<g, blk, 0, stream>>>(A, aO, aI, lda, Bt, bO, bI, ldb,
                                              C, cO, cI, ldc, M, N, K, zInner,
                                              bias, res, scale, flags);
    };

    // -- weight conversion (f32 [K,N] -> bf16 [N,K]) --
    for (int l = 0; l < Ll; ++l) {
        size_t w = (size_t)l * Dm * Dm;
        convert_transpose<<<cdiv((long long)Dm * Dm, 256), blk, 0, stream>>>(Wq + w, WqT + w, Dm, Dm);
        convert_transpose<<<cdiv((long long)Dm * Dm, 256), blk, 0, stream>>>(Wk + w, WkT + w, Dm, Dm);
        convert_transpose<<<cdiv((long long)Dm * Dm, 256), blk, 0, stream>>>(Wv + w, WvT + w, Dm, Dm);
        convert_transpose<<<cdiv((long long)Dm * Dm, 256), blk, 0, stream>>>(Wo + w, WoT + w, Dm, Dm);
        size_t wf = (size_t)l * Dm * FFf;
        convert_transpose<<<cdiv((long long)Dm * FFf, 256), blk, 0, stream>>>(W1 + wf, W1T + wf, Dm, FFf);
        convert_transpose<<<cdiv((long long)Dm * FFf, 256), blk, 0, stream>>>(W2 + wf, W2T + wf, FFf, Dm);
    }
    convert_transpose<<<cdiv((long long)Dm * Vv, 256), blk, 0, stream>>>(lmh, lmhT, Dm, Vv);

    // -- embeddings --
    embed_kernel<<<cdiv((long long)Mm * Dm, 256), blk, 0, stream>>>(idx, tok, pos, x, Mm * Dm);

    // -- transformer blocks --
    for (int l = 0; l < Ll; ++l) {
        const size_t wDD = (size_t)l * Dm * Dm, wDF = (size_t)l * Dm * FFf;
        layernorm_bf16<<<Mm, blk, 0, stream>>>(x, ln1g + l * Dm, ln1b + l * Dm, hbf);
        gemm(hbf, 0, 0, Dm, WqT + wDD, 0, 0, Dm, qbf, 0, 0, Dm,
             Mm, Dm, Dm, 1, 1, nullptr, nullptr, 1.f, F_OUT_BF16);
        gemm(hbf, 0, 0, Dm, WkT + wDD, 0, 0, Dm, kbf, 0, 0, Dm,
             Mm, Dm, Dm, 1, 1, nullptr, nullptr, 1.f, F_OUT_BF16);
        gemm(hbf, 0, 0, Dm, WvT + wDD, 0, 0, Dm, vtbf, 0, 0, Mm,   // transposed store -> Vᵀ
             Mm, Dm, Dm, 1, 1, nullptr, nullptr, 1.f, F_OUT_BF16 | F_TRANS);
        // scores[b,h,q,k] = (Q·Kᵀ)/8  — per-(b,h) batched, causal tiles skipped
        gemm(qbf, (long long)Ss * Dm, DHd, Dm,
             kbf, (long long)Ss * Dm, DHd, Dm,
             sc,  (long long)Hh * Ss * Ss, (long long)Ss * Ss, Ss,
             Ss, Ss, DHd, Bb * Hh, Hh, nullptr, nullptr, 0.125f, F_CAUSAL);
        softmax_causal<<<Bb * Hh * Ss, blk, 0, stream>>>(sc, att);
        // O = P·V  (B operand = Vᵀ rows, contiguous in K); K clipped causally,
        // narrow wave layout since N=64
        gemm(att, (long long)Hh * Ss * Ss, (long long)Ss * Ss, Ss,
             vtbf, (long long)Ss, (long long)DHd * Mm, Mm,
             obf, (long long)Ss * Dm, DHd, Dm,
             Ss, DHd, Ss, Bb * Hh, Hh, nullptr, nullptr, 1.f,
             F_OUT_BF16 | F_KCAUSAL | F_NARROW);
        gemm(obf, 0, 0, Dm, WoT + wDD, 0, 0, Dm, x, 0, 0, Dm,
             Mm, Dm, Dm, 1, 1, nullptr, x, 1.f, 0);                // x += O·Wo
        layernorm_bf16<<<Mm, blk, 0, stream>>>(x, ln2g + l * Dm, ln2b + l * Dm, hbf);
        gemm(hbf, 0, 0, Dm, W1T + wDF, 0, 0, Dm, ffbf, 0, 0, FFf,
             Mm, FFf, Dm, 1, 1, b1 + (size_t)l * FFf, nullptr, 1.f, F_OUT_BF16 | F_RELU);
        gemm(ffbf, 0, 0, FFf, W2T + wDF, 0, 0, FFf, x, 0, 0, Dm,
             Mm, Dm, FFf, 1, 1, b2 + (size_t)l * Dm, x, 1.f, 0);   // x += relu(..)·W2 + b2
    }

    // -- final LN + lm_head + loss --
    layernorm_bf16<<<Mm, blk, 0, stream>>>(x, lnfg, lnfb, hbf);
    gemm(hbf, 0, 0, Dm, lmhT, 0, 0, Dm, logits, 0, 0, Vv,
         Mm, Vv, Dm, 1, 1, nullptr, nullptr, 1.f, 0);
    loss_rows<<<Mm, blk, 0, stream>>>(logits, tgt, nll);
    loss_mean<<<1, blk, 0, stream>>>(nll, lossOut, Mm);
}
